// GRUDecoder_46256797778270
// MI455X (gfx1250) — matrix operand order repaired and checked by reading, not compiled, not run on hardware
//
#include <hip/hip_runtime.h>

#define DEV __device__ __forceinline__

typedef __attribute__((ext_vector_type(16))) __bf16 v16bf;
typedef __attribute__((ext_vector_type(8)))  float  v8f;
typedef __attribute__((ext_vector_type(4)))  unsigned int vu4;

// Problem constants (from reference): B=256, T=512, H=512, E=256, V=1024
constexpr int B   = 256;
constexpr int T   = 512;
constexpr int H   = 512;
constexpr int E   = 256;
constexpr int V   = 1024;
constexpr int G   = 3 * H;      // 1536 gate width
constexpr int EH  = E + H;      // 768  w_ih row stride
constexpr int EHH = E + 2 * H;  // 1280 w_out row stride
constexpr int TC  = 16;         // time-chunk for the batched output projection

// ---------------- helpers ----------------

DEV unsigned short f2bf(float x) {
  unsigned int u = __builtin_bit_cast(unsigned int, x);
  u += 0x7FFFu + ((u >> 16) & 1u);            // round-to-nearest-even
  return (unsigned short)(u >> 16);
}

union Frag { v16bf v; vu4 q[2]; };

// A-operand fragment loaded straight from a row-major bf16 row (16x32 tile,
// one matrix row per lane): per-lane data = two contiguous 16B segments.
DEV v16bf load_a_row(const unsigned short* __restrict__ row, int k0, int lane) {
  const int kb = (lane & 16) ? 8 : 0;
  Frag f;
  f.q[0] = *(const vu4*)(row + k0 + kb);
  f.q[1] = *(const vu4*)(row + k0 + kb + 16);
  return f.v;
}

// B-operand fragment loaded straight from a row-major bf16 row (32x16 tile,
// one matrix column per lane; lane's K = kb..kb+7 and 16+kb..16+kb+7, kb = 8h).
DEV v16bf load_b_row(const unsigned short* __restrict__ row, int k0, int lane) {
  const int kb = (lane & 16) ? 8 : 0;
  Frag f;
  f.q[0] = *(const vu4*)(row + k0 + kb);
  f.q[1] = *(const vu4*)(row + k0 + kb + 16);
  return f.v;
}

// Fragment from pre-packed weights: each lane's 8 dwords stored contiguously.
DEV v16bf load_pack(const unsigned int* __restrict__ pack, int tile, int lane) {
  const vu4* p = (const vu4*)(pack + (size_t)tile * 256 + lane * 8);
  Frag f;
  f.q[0] = p[0];
  f.q[1] = p[1];
  return f.v;
}

DEV v8f wmma_bf16(v16bf a, v16bf b, v8f c) {
  return __builtin_amdgcn_wmma_f32_16x16x32_bf16(false, a, false, b, (short)0, c,
                                                 false, false);
}

// ---------------- setup kernels ----------------

// Pack W as B-operand (B[k][n] = W[n][colOff+k]) in per-lane fragment order:
// n = lane&15, kb = (lane&16 ? 8 : 0), K = p<4 ? kb+2p : 16+kb+2(p-4).
__global__ __launch_bounds__(256) void pack_b_kernel(const float* __restrict__ W,
                                                     int rowStride, int colOff,
                                                     int KT, int NT,
                                                     unsigned int* __restrict__ dst) {
  int id = blockIdx.x * 256 + threadIdx.x;
  if (id >= KT * NT * 256) return;
  int p    = id & 7;
  int lane = (id >> 3) & 31;
  int tile = id >> 8;
  int nt = tile % NT, kt = tile / NT;
  int n  = nt * 16 + (lane & 15);
  int kb = (lane & 16) ? 8 : 0;
  int k  = kt * 32 + ((p < 4) ? (kb + 2 * p) : (16 + kb + 2 * (p - 4)));
  const float* s = W + (size_t)n * rowStride + colOff + k;
  dst[id] = (unsigned int)f2bf(s[0]) | ((unsigned int)f2bf(s[1]) << 16);
}

// Pack W as A-operand (A[m][k] = W[m][colOff+k]) in per-lane fragment order:
// m = lane&15, kb = (lane&16 ? 8 : 0), K = p<4 ? kb+2p : 16+kb+2(p-4).
__global__ __launch_bounds__(256) void pack_a_kernel(const float* __restrict__ W,
                                                     int rowStride, int colOff,
                                                     int KT, int MT,
                                                     unsigned int* __restrict__ dst) {
  int id = blockIdx.x * 256 + threadIdx.x;
  if (id >= KT * MT * 256) return;
  int p    = id & 7;
  int lane = (id >> 3) & 31;
  int tile = id >> 8;
  int mt = tile % MT, kt = tile / MT;
  int m  = mt * 16 + (lane & 15);
  int kb = (lane & 16) ? 8 : 0;
  int k  = kt * 32 + ((p < 4) ? (kb + 2 * p) : (16 + kb + 2 * (p - 4)));
  const float* s = W + (size_t)m * rowStride + colOff + k;
  dst[id] = (unsigned int)f2bf(s[0]) | ((unsigned int)f2bf(s[1]) << 16);
}

__global__ __launch_bounds__(256) void conv_emb_kernel(const float* __restrict__ src,
                                                       unsigned short* __restrict__ dst) {
  int id = blockIdx.x * 256 + threadIdx.x;   // V*E total
  dst[id] = f2bf(src[id]);
}

__global__ __launch_bounds__(256) void init_h_kernel(const float* __restrict__ hidden,
                                                     float* __restrict__ hf,
                                                     unsigned short* __restrict__ hb) {
  int id = blockIdx.x * 256 + threadIdx.x;   // B*H total
  float v = hidden[id];
  hf[id] = v;
  hb[id] = f2bf(v);
}

// gxc[b][g] = b_ih[g] + ctx[b] . w_ih[g][E:E+H]   (one-time, f32)
__global__ __launch_bounds__(256) void gxc_kernel(const float* __restrict__ ctx,
                                                  const float* __restrict__ w_ih,
                                                  const float* __restrict__ b_ih,
                                                  float* __restrict__ gxc) {
  int id = blockIdx.x * 256 + threadIdx.x;
  if (id >= B * G) return;
  int b = id / G, g = id % G;
  const float* c = ctx + (size_t)b * H;
  const float* w = w_ih + (size_t)g * EH + E;
  float s = b_ih[g];
  for (int k = 0; k < H; ++k) s = fmaf(c[k], w[k], s);
  gxc[id] = s;
}

// outc[b][v] = b_out[v] + ctx[b] . w_out[v][E+H:]   (one-time, f32)
__global__ __launch_bounds__(256) void outc_kernel(const float* __restrict__ ctx,
                                                   const float* __restrict__ w_out,
                                                   const float* __restrict__ b_out,
                                                   float* __restrict__ outc) {
  int id = blockIdx.x * 256 + threadIdx.x;
  if (id >= B * V) return;
  int b = id / V, v = id % V;
  const float* c = ctx + (size_t)b * H;
  const float* w = w_out + (size_t)v * EHH + E + H;
  float s = b_out[v];
  for (int k = 0; k < H; ++k) s = fmaf(c[k], w[k], s);
  outc[id] = s;
}

// ---------------- WMMA GEMM kernels ----------------

// One recurrence step: gates + h update. Split-K: each 16x16 (b,j) tile is
// computed by TWO waves (K-halves), halving the dependent load->WMMA chain on
// the serial critical path and doubling WGP coverage (128 WGs). Partial
// accumulators are combined through LDS; the lower wave finishes gate math.
__global__ __launch_bounds__(256) void step_kernel(const int* __restrict__ target,
                                                   const unsigned short* __restrict__ embB,
                                                   const unsigned int* __restrict__ packWihx,
                                                   const unsigned int* __restrict__ packWhh,
                                                   const float* __restrict__ gxc,
                                                   const float* __restrict__ b_hh,
                                                   const float* __restrict__ hf_prev,
                                                   const unsigned short* __restrict__ hbf_prev,
                                                   float* __restrict__ hf_next,
                                                   unsigned short* __restrict__ hbf_next,
                                                   int t) {
  __shared__ float red[4][32][33];         // padded: conflict-free lane-major access

  const int lane = threadIdx.x & 31;
  const int wave = threadIdx.x >> 5;
  const int pair = wave >> 1;              // tile slot within WG: 0..3
  const int kh   = wave & 1;               // K-half: 0 = low, 1 = high
  const int tile = blockIdx.x * 4 + pair;  // 0..511
  const int bt = tile >> 5;                // 16 b-tiles
  const int jt = tile & 31;                // 32 j-tiles (H/16)
  const int b0 = bt * 16, j0 = jt * 16;

  const int arow = b0 + (lane & 15);
  const int xi   = target[arow * (T + 1) + t];
  const unsigned short* xRow = embB + (size_t)xi * E;
  const unsigned short* hRow = hbf_prev + (size_t)arow * H;

  v8f accR = {}, accZ = {}, accXN = {}, accHN = {};

  const int xs = kh * (E / 64);            // 4-iteration x half
  const int hs = kh * (H / 64);            // 8-iteration h half

  // x-part: gx_{r,z,n} += x . w_ih[:, 0:E]   (this wave's K-half)
#pragma unroll
  for (int i = 0; i < E / 64; ++i) {
    const int kt = xs + i;
    v16bf a  = load_a_row(xRow, kt * 32, lane);
    v16bf bR = load_pack(packWihx, kt * 96 + jt,      lane);
    v16bf bZ = load_pack(packWihx, kt * 96 + jt + 32, lane);
    v16bf bN = load_pack(packWihx, kt * 96 + jt + 64, lane);
    accR  = wmma_bf16(a, bR, accR);
    accZ  = wmma_bf16(a, bZ, accZ);
    accXN = wmma_bf16(a, bN, accXN);
  }
  // h-part: gh_{r,z,n} += h . w_hh   (this wave's K-half; r,z fused)
#pragma unroll
  for (int i = 0; i < H / 64; ++i) {
    const int kt = hs + i;
    v16bf a  = load_a_row(hRow, kt * 32, lane);
    v16bf bR = load_pack(packWhh, kt * 96 + jt,      lane);
    v16bf bZ = load_pack(packWhh, kt * 96 + jt + 32, lane);
    v16bf bN = load_pack(packWhh, kt * 96 + jt + 64, lane);
    accR  = wmma_bf16(a, bR, accR);
    accZ  = wmma_bf16(a, bZ, accZ);
    accHN = wmma_bf16(a, bN, accHN);
  }

  // Combine K-halves through LDS (wave-uniform branches keep EXEC all-ones)
  if (kh) {
    float* p = &red[pair][lane][0];
#pragma unroll
    for (int r = 0; r < 8; ++r) {
      p[r]      = accR[r];
      p[8 + r]  = accZ[r];
      p[16 + r] = accXN[r];
      p[24 + r] = accHN[r];
    }
  }
  __syncthreads();
  if (kh) return;
  {
    const float* p = &red[pair][lane][0];
#pragma unroll
    for (int r = 0; r < 8; ++r) {
      accR[r]  += p[r];
      accZ[r]  += p[8 + r];
      accXN[r] += p[16 + r];
      accHN[r] += p[24 + r];
    }
  }

  const int n  = lane & 15;
  const int mb = (lane & 16) ? 8 : 0;
#pragma unroll
  for (int r = 0; r < 8; ++r) {
    int bb = b0 + mb + r;
    int jj = j0 + n;
    const float* gx = gxc + (size_t)bb * G;
    float pr = accR[r]  + gx[jj]         + b_hh[jj];
    float pz = accZ[r]  + gx[jj + H]     + b_hh[jj + H];
    float gn = accXN[r] + gx[jj + 2 * H];
    float hn = accHN[r] + b_hh[jj + 2 * H];
    float rg = 1.0f / (1.0f + __expf(-pr));
    float zg = 1.0f / (1.0f + __expf(-pz));
    float nn = tanhf(gn + rg * hn);
    float ho = hf_prev[(size_t)bb * H + jj];
    float hv = (1.0f - zg) * nn + zg * ho;
    hf_next[(size_t)bb * H + jj]  = hv;
    hbf_next[(size_t)bb * H + jj] = f2bf(hv);
  }
}

// Batched output projection for a 16-step chunk, role-swapped GEMM:
// M = v (weights as A operand), N = t (16 lanes = 16 consecutive timesteps),
// K = feature. out[b][v][t] = outc[b][v] + x_t . w_out[v][0:E] + h_t . w_out[v][E:E+H]
// Stores are lane-coalesced along t; d_out is written exactly once, no RMW.
__global__ __launch_bounds__(256) void out_chunk_kernel(const int* __restrict__ target,
                                                        const unsigned short* __restrict__ embB,
                                                        const unsigned int* __restrict__ packWoxA,
                                                        const unsigned int* __restrict__ packWohA,
                                                        const float* __restrict__ outc,
                                                        const unsigned short* __restrict__ hbf_all,
                                                        float* __restrict__ out, int t0) {
  const int lane = threadIdx.x & 31;
  const int wave = threadIdx.x >> 5;
  const int wid  = blockIdx.x * 8 + wave;  // B * V/16 = 16384 waves
  const int b    = wid >> 6;               // 0..255
  const int vt   = wid & 63;               // 0..63
  const int v0   = vt * 16;

  // Per-lane B-operand rows: lane's column n == timestep t0 + (lane&15)
  const int tt = lane & 15;
  const int xi = target[b * (T + 1) + t0 + tt];
  const unsigned short* xRow = embB + (size_t)xi * E;
  const unsigned short* hRow = hbf_all + ((size_t)tt * B + b) * H;  // slice tt = h at step t0+tt

  v8f acc = {};
#pragma unroll
  for (int kt = 0; kt < E / 32; ++kt) {
    v16bf a  = load_pack(packWoxA, kt * 64 + vt, lane);
    v16bf bb = load_b_row(xRow, kt * 32, lane);
    acc = wmma_bf16(a, bb, acc);
  }
#pragma unroll
  for (int kt = 0; kt < H / 32; ++kt) {
    v16bf a  = load_pack(packWohA, kt * 64 + vt, lane);
    v16bf bb = load_b_row(hRow, kt * 32, lane);
    acc = wmma_bf16(a, bb, acc);
  }

  const int n  = lane & 15;                // timestep within chunk
  const int mb = (lane & 16) ? 8 : 0;      // v row offset
#pragma unroll
  for (int r = 0; r < 8; ++r) {
    int vv = v0 + mb + r;
    out[((size_t)b * V + vv) * T + t0 + n] = acc[r] + outc[(size_t)b * V + vv];
  }
}

// ---------------- host launch ----------------

extern "C" void kernel_launch(void* const* d_in, const int* in_sizes, int n_in,
                              void* d_out, int out_size, void* d_ws, size_t ws_size,
                              hipStream_t stream) {
  (void)in_sizes; (void)n_in; (void)out_size; (void)ws_size;

  const int*   target = (const int*)d_in[0];
  const float* hidden = (const float*)d_in[1];
  const float* context= (const float*)d_in[2];
  const float* emb    = (const float*)d_in[3];
  const float* w_ih   = (const float*)d_in[4];
  const float* w_hh   = (const float*)d_in[5];
  const float* b_ih   = (const float*)d_in[6];
  const float* b_hh   = (const float*)d_in[7];
  const float* w_out  = (const float*)d_in[8];
  const float* b_out  = (const float*)d_in[9];
  float* out = (float*)d_out;

  // Workspace carve-out (~13 MB total)
  char* ws = (char*)d_ws;
  size_t off = 0;
  auto take = [&](size_t bytes) -> char* {
    char* p = ws + off;
    off = (off + bytes + 255) & ~(size_t)255;
    return p;
  };
  unsigned short* embB    = (unsigned short*)take((size_t)V * E * 2);
  float*          hf0     = (float*)take((size_t)B * H * 4);
  float*          hf1     = (float*)take((size_t)B * H * 4);
  unsigned short* hbInit  = (unsigned short*)take((size_t)B * H * 2);
  unsigned short* hbfAll  = (unsigned short*)take((size_t)TC * B * H * 2);  // 16-step ring
  float*          gxc     = (float*)take((size_t)B * G * 4);
  float*          outc    = (float*)take((size_t)B * V * 4);
  unsigned int*   pIhx    = (unsigned int*)take((size_t)8  * 96 * 256 * 4);  // B-packed
  unsigned int*   pHh     = (unsigned int*)take((size_t)16 * 96 * 256 * 4);  // B-packed
  unsigned int*   pAOx    = (unsigned int*)take((size_t)8  * 64 * 256 * 4);  // A-packed
  unsigned int*   pAOh    = (unsigned int*)take((size_t)16 * 64 * 256 * 4);  // A-packed

  // One-time conversions / precomputation (per launch: deterministic)
  pack_b_kernel<<<(8 * 96 * 256) / 256, 256, 0, stream>>>(w_ih,  EH,  0, 8,  96, pIhx);
  pack_b_kernel<<<(16 * 96 * 256) / 256, 256, 0, stream>>>(w_hh, H,   0, 16, 96, pHh);
  pack_a_kernel<<<(8 * 64 * 256) / 256, 256, 0, stream>>>(w_out, EHH, 0, 8,  64, pAOx);
  pack_a_kernel<<<(16 * 64 * 256) / 256, 256, 0, stream>>>(w_out, EHH, E, 16, 64, pAOh);
  conv_emb_kernel<<<(V * E) / 256, 256, 0, stream>>>(emb, embB);
  init_h_kernel<<<(B * H) / 256, 256, 0, stream>>>(hidden, hf0, hbInit);
  gxc_kernel<<<(B * G) / 256, 256, 0, stream>>>(context, w_ih, b_ih, gxc);
  outc_kernel<<<(B * V) / 256, 256, 0, stream>>>(context, w_out, b_out, outc);

  // Sequential recurrence; every 16 steps, one batched, coalesced output GEMM.
  float* hfp[2] = {hf0, hf1};
  for (int t = 0; t < T; ++t) {
    int cur = t & 1, nxt = cur ^ 1;
    const unsigned short* hbPrev =
        (t == 0) ? hbInit : (hbfAll + (size_t)((t - 1) & (TC - 1)) * B * H);
    unsigned short* hbNext = hbfAll + (size_t)(t & (TC - 1)) * B * H;
    step_kernel<<<128, 256, 0, stream>>>(target, embB, pIhx, pHh, gxc, b_hh,
                                         hfp[cur], hbPrev, hfp[nxt], hbNext, t);
    if ((t & (TC - 1)) == TC - 1) {
      out_chunk_kernel<<<(B * V / 16) / 8, 256, 0, stream>>>(
          target, embB, pAOx, pAOh, outc, hbfAll, out, t - (TC - 1));
    }
  }
}
